// AttentionLayer_56684978372683
// MI455X (gfx1250) — compile-verified, matrix operand order checked by reading, not run on hardware
//
#include <hip/hip_runtime.h>
#include <hip/hip_bf16.h>

// ---------------------------------------------------------------------------
// CDNA5 (gfx1250) bf16-WMMA attention layer.
//   q = (X Wq + bq) * 1/sqrt(hd) ; k,v = X W + b   (bf16 WMMA GEMM, f32 accum)
//   flash attention per (b, h)                     (bf16 WMMA, online softmax)
//   out = ctx Wo + bo                              (bf16 WMMA GEMM, f32 out)
// Both hot loops are software-pipelined with double-buffered LDS slabs:
// next chunk's global->LDS movement overlaps the current chunk's WMMAs,
// one workgroup barrier per iteration.
// ---------------------------------------------------------------------------

typedef __attribute__((ext_vector_type(16))) __bf16 bf16x16;
typedef __attribute__((ext_vector_type(8)))  float  f32x8;

union AFrag { bf16x16 v; unsigned int u[8]; unsigned short s[16]; };
union CFrag { f32x8   v; float f[8]; };
union Pack8 { uint4 q; unsigned short s[8]; };

#define BATCH 4
#define SEQ   2048
#define DMODEL 1024
#define NHEAD 16
#define HDIM  64

static __device__ __forceinline__ f32x8 wmma_bf16(bf16x16 a, bf16x16 b, f32x8 c) {
  // D = A(16x32 bf16) * B(32x16 bf16) + C(16x16 f32)
  return __builtin_amdgcn_wmma_f32_16x16x32_bf16(false, a, false, b, (short)0, c,
                                                 false, false);
}

// 16-bit A/B fragment K-base for register j (ISA 7.12.2):
// lanes 0-15 hold K = {0..7, 16..23}; lanes 16-31 hold K = {8..15, 24..31}
static __device__ __forceinline__ int kbase16(int j, int hi) {
  return (j < 4 ? 2 * j : 16 + 2 * (j - 4)) + (hi ? 8 : 0);
}

static __device__ __forceinline__ unsigned short f32_to_bf16(float x) {
  unsigned int u = __float_as_uint(x);
  unsigned int r = u + 0x7FFFu + ((u >> 16) & 1u);  // round-to-nearest-even
  return (unsigned short)(r >> 16);
}

static __device__ __forceinline__ void wait_asynccnt0() {
#if __has_builtin(__builtin_amdgcn_s_wait_asynccnt)
  __builtin_amdgcn_s_wait_asynccnt(0);
#else
  asm volatile("s_wait_asynccnt 0" ::: "memory");
#endif
}

// Async DMA of 16 bytes global -> LDS (CDNA5 GLOBAL_LOAD_ASYNC_TO_LDS_B128).
static __device__ __forceinline__ void async_g2l_b128(unsigned lds_byte_addr,
                                                      const void* gptr) {
  asm volatile("global_load_async_to_lds_b128 %0, %1, off"
               :: "v"(lds_byte_addr), "v"((unsigned long long)(size_t)gptr)
               : "memory");
}

// ---- 16-lane row reductions: DPP fused directly into the combining op ------
// dst = op(dpp(src0), src1) -> one VALU per hop, no movs, no canonicalize.
#define DPP_MAX_HOP(v, CTRL)                                              \
  asm("v_max_num_f32_dpp %0, %1, %0 " CTRL                                \
      " row_mask:0xf bank_mask:0xf bound_ctrl:1"                          \
      : "+v"(v) : "v"(v))
#define DPP_ADD_HOP(v, CTRL)                                              \
  asm("v_add_f32_dpp %0, %1, %0 " CTRL                                    \
      " row_mask:0xf bank_mask:0xf bound_ctrl:1"                          \
      : "+v"(v) : "v"(v))

static __device__ __forceinline__ float red16_max(float v) {
  DPP_MAX_HOP(v, "quad_perm:[1,0,3,2]");  // xor 1
  DPP_MAX_HOP(v, "quad_perm:[2,3,0,1]");  // xor 2
  DPP_MAX_HOP(v, "row_half_mirror");      // xor 7
  DPP_MAX_HOP(v, "row_mirror");           // xor 15
  return v;
}
static __device__ __forceinline__ float red16_sum(float v) {
  DPP_ADD_HOP(v, "quad_perm:[1,0,3,2]");
  DPP_ADD_HOP(v, "quad_perm:[2,3,0,1]");
  DPP_ADD_HOP(v, "row_half_mirror");
  DPP_ADD_HOP(v, "row_mirror");
  return v;
}

// ---------------------------------------------------------------------------
// fp32 -> bf16 conversion (grid-stride)
// ---------------------------------------------------------------------------
__global__ void cvt_f32_bf16(const float* __restrict__ src,
                             unsigned short* __restrict__ dst, int n) {
  int i = blockIdx.x * blockDim.x + threadIdx.x;
  int stride = gridDim.x * blockDim.x;
  for (; i < n; i += stride) dst[i] = f32_to_bf16(src[i]);
}

// ---------------------------------------------------------------------------
// bf16 GEMM:  out[M,1024] = ((A[M,1024] * W[1024,1024]) + bias) * oscale
// Workgroup: 256 threads (8 waves). Block tile 256(M) x 64(N).
// Each wave owns a 32x64 tile: two 16-row subtiles sharing every B fragment.
// Double-buffered W K-slab: stage slab k0+32 while WMMAs consume slab k0.
// ---------------------------------------------------------------------------
template <int F32OUT>
__global__ void __launch_bounds__(256)
gemm_bf16(const unsigned short* __restrict__ A,
          const unsigned short* __restrict__ W,
          const float* __restrict__ bias,
          unsigned short* __restrict__ outB,
          float* __restrict__ outF,
          float oscale) {
  __shared__ unsigned short ldsWt[2][64 * 32];  // [buf][n][k] transposed slabs

  const int tid    = threadIdx.x;
  const int wave   = tid >> 5;
  const int lane   = tid & 31;
  const int lane16 = lane & 15;
  const int hi     = lane >> 4;

  const int n0 = blockIdx.x * 64;
  const int m0 = blockIdx.y * 256 + wave * 32;

  // staging geometry: thread covers W[k0+wk][n0+wn8 .. n0+wn8+7]
  const int wk  = tid >> 3;        // 0..31
  const int wn8 = (tid & 7) * 8;   // 0..56

  auto stageW = [&](int buf, int k0) {
    Pack8 wv;
    wv.q = *(const uint4*)(W + ((k0 + wk) << 10) + n0 + wn8);
#pragma unroll
    for (int j = 0; j < 8; ++j) ldsWt[buf][(wn8 + j) * 32 + wk] = wv.s[j];
  };

  CFrag acc[2][4];
#pragma unroll
  for (int s = 0; s < 2; ++s)
#pragma unroll
    for (int nt = 0; nt < 4; ++nt)
#pragma unroll
      for (int c = 0; c < 8; ++c) acc[s][nt].f[c] = 0.0f;

  const unsigned short* arow0 = A + ((m0 + lane16) << 10);
  const unsigned short* arow1 = arow0 + (16 << 10);

  stageW(0, 0);
  __syncthreads();

  int buf = 0;
  for (int k0 = 0; k0 < DMODEL; k0 += 32) {
    if (k0 + 32 < DMODEL) stageW(buf ^ 1, k0 + 32);  // overlaps with WMMAs below

    // two A fragments (16x32 each) straight from global, packed-pair loads
    AFrag a0, a1;
#pragma unroll
    for (int j = 0; j < 8; ++j) {
      int kb = k0 + kbase16(j, hi);
      a0.u[j] = *(const unsigned int*)(arow0 + kb);
      a1.u[j] = *(const unsigned int*)(arow1 + kb);
    }

#pragma unroll
    for (int nt = 0; nt < 4; ++nt) {
      AFrag b;
#pragma unroll
      for (int j = 0; j < 8; ++j)
        b.u[j] = *(const unsigned int*)&ldsWt[buf][(nt * 16 + lane16) * 32 +
                                                   kbase16(j, hi)];
      acc[0][nt].v = wmma_bf16(a0.v, b.v, acc[0][nt].v);
      acc[1][nt].v = wmma_bf16(a1.v, b.v, acc[1][nt].v);
    }

    __syncthreads();   // single barrier per k-step
    buf ^= 1;
  }

  // epilogue: (acc + bias) * oscale, store (C layout: row = c+8*hi, col = lane16)
#pragma unroll
  for (int s = 0; s < 2; ++s) {
    int mbase = m0 + s * 16 + hi * 8;
#pragma unroll
    for (int nt = 0; nt < 4; ++nt) {
      int n = n0 + nt * 16 + lane16;
      float bz = bias[n];
#pragma unroll
      for (int c = 0; c < 8; ++c) {
        int idx = ((mbase + c) << 10) + n;
        float val = (acc[s][nt].f[c] + bz) * oscale;
        if (F32OUT) outF[idx] = val;
        else        outB[idx] = f32_to_bf16(val);
      }
    }
  }
}

// ---------------------------------------------------------------------------
// Flash attention: one workgroup = one (b, h) and 128 query rows (16 per wave).
// S iterated in chunks of 32 keys, double-buffered: chunk i+1's K async-DMA
// and V transpose-staging are issued before computing chunk i.
// Q is pre-scaled by 1/sqrt(hd) in its projection epilogue.
// ---------------------------------------------------------------------------
__global__ void __launch_bounds__(256)
flash_attn(const unsigned short* __restrict__ Q,
           const unsigned short* __restrict__ Kb,
           const unsigned short* __restrict__ Vb,
           unsigned short* __restrict__ ctx) {
  __shared__ unsigned short ldsK[2][32 * 64];   // [buf][key][dim]
  __shared__ unsigned short ldsVt[2][64 * 32];  // [buf][dim][key]
  __shared__ unsigned short ldsP[8 * 16 * 32];  // per-wave P tile (16x32 bf16)

  const int tid    = threadIdx.x;
  const int wave   = tid >> 5;
  const int lane   = tid & 31;
  const int lane16 = lane & 15;
  const int hi     = lane >> 4;

  const int h  = blockIdx.y;
  const int b  = blockIdx.z;
  const int m0 = blockIdx.x * 128 + wave * 16;

  const int rowQ = ((b * SEQ + m0 + lane16) << 10) + h * HDIM;

  // Q fragments for K-dim chunks 0..31 and 32..63 (hd = 64)
  AFrag aq[2];
#pragma unroll
  for (int f = 0; f < 2; ++f)
#pragma unroll
    for (int j = 0; j < 8; ++j)
      aq[f].u[j] = *(const unsigned int*)(Q + rowQ + f * 32 + kbase16(j, hi));

  float mrow[8], lrow[8];
  CFrag acc[4];
#pragma unroll
  for (int c = 0; c < 8; ++c) { mrow[c] = -3.0e38f; lrow[c] = 0.0f; }
#pragma unroll
  for (int nt = 0; nt < 4; ++nt)
#pragma unroll
    for (int c = 0; c < 8; ++c) acc[nt].f[c] = 0.0f;

  const unsigned short* pbase = ldsP + wave * 512;

  // per-thread staging geometry: 16 bytes (8 bf16) per thread, 4KB per chunk
  const int kkey = tid >> 3;            // 0..31 (row within chunk)
  const int kcol = (tid & 7) * 8;       // 0..56 (dim base)
  const int kvoff = ((b * SEQ + kkey) << 10) + h * HDIM + kcol;

  auto stageK = [&](int buf, int s0) {  // async DMA, one b128 per thread
    async_g2l_b128((unsigned)(size_t)&ldsK[buf][kkey * 64 + kcol],
                   Kb + kvoff + (s0 << 10));
  };
  auto stageV = [&](int buf, int s0) {  // b128 load + strided transpose stores
    Pack8 vv;
    vv.q = *(const uint4*)(Vb + kvoff + (s0 << 10));
#pragma unroll
    for (int j = 0; j < 8; ++j) ldsVt[buf][(kcol + j) * 32 + kkey] = vv.s[j];
  };

  stageK(0, 0);
  stageV(0, 0);
  wait_asynccnt0();
  __syncthreads();

  int cur = 0;
  for (int s0 = 0; s0 < SEQ; s0 += 32) {
    if (s0 + 32 < SEQ) {
      stageK(cur ^ 1, s0 + 32);   // overlaps with the whole chunk body below
      stageV(cur ^ 1, s0 + 32);
      if (s0 + 64 < SEQ)          // warm L2 beyond the software pipeline
        __builtin_prefetch(Kb + kvoff + ((s0 + 64) << 10), 0, 1);
    }

    // scores: two 16x16 n-tiles over 32 keys, contraction hd = 64
    CFrag sf[2];
#pragma unroll
    for (int nt2 = 0; nt2 < 2; ++nt2) {
      f32x8 s;
#pragma unroll
      for (int c = 0; c < 8; ++c) s[c] = 0.0f;
#pragma unroll
      for (int f = 0; f < 2; ++f) {
        AFrag bk;
#pragma unroll
        for (int j = 0; j < 8; ++j)
          bk.u[j] = *(const unsigned int*)&ldsK[cur][(nt2 * 16 + lane16) * 64 +
                                                     f * 32 + kbase16(j, hi)];
        s = wmma_bf16(aq[f].v, bk.v, s);
      }
      sf[nt2].v = s;
    }

    // online softmax: fused-DPP row reductions across the 16-lane groups
    float mnew[8], alpha[8];
#pragma unroll
    for (int c = 0; c < 8; ++c) {
      float mx = red16_max(fmaxf(sf[0].f[c], sf[1].f[c]));
      mnew[c]  = fmaxf(mrow[c], mx);
      alpha[c] = __expf(mrow[c] - mnew[c]);
    }
#pragma unroll
    for (int nt2 = 0; nt2 < 2; ++nt2)
#pragma unroll
      for (int c = 0; c < 8; ++c)
        sf[nt2].f[c] = __expf(sf[nt2].f[c] - mnew[c]);
#pragma unroll
    for (int c = 0; c < 8; ++c) {
      float sum = red16_sum(sf[0].f[c] + sf[1].f[c]);
      lrow[c] = lrow[c] * alpha[c] + sum;
      mrow[c] = mnew[c];
    }
#pragma unroll
    for (int nt = 0; nt < 4; ++nt)
#pragma unroll
      for (int c = 0; c < 8; ++c) acc[nt].f[c] *= alpha[c];

    // P: C layout -> per-wave LDS -> reload in A layout (wave-local, DS in order)
#pragma unroll
    for (int nt2 = 0; nt2 < 2; ++nt2)
#pragma unroll
      for (int c = 0; c < 8; ++c) {
        int row = c + hi * 8;
        ((unsigned short*)pbase)[row * 32 + nt2 * 16 + lane16] =
            f32_to_bf16(sf[nt2].f[c]);
      }
    AFrag pa;
#pragma unroll
    for (int j = 0; j < 8; ++j)
      pa.u[j] = *(const unsigned int*)&pbase[lane16 * 32 + kbase16(j, hi)];

#pragma unroll
    for (int nt = 0; nt < 4; ++nt) {
      AFrag bv;
#pragma unroll
      for (int j = 0; j < 8; ++j)
        bv.u[j] = *(const unsigned int*)&ldsVt[cur][(nt * 16 + lane16) * 32 +
                                                    kbase16(j, hi)];
      acc[nt].v = wmma_bf16(pa.v, bv.v, acc[nt].v);
    }

    wait_asynccnt0();   // next chunk's DMA done
    __syncthreads();    // single barrier per chunk
    cur ^= 1;
  }

  // normalize and write context (bf16)
#pragma unroll
  for (int c = 0; c < 8; ++c) {
    float inv = 1.0f / lrow[c];
    int m = m0 + c + hi * 8;
#pragma unroll
    for (int nt = 0; nt < 4; ++nt)
      ctx[((b * SEQ + m) << 10) + h * HDIM + nt * 16 + lane16] =
          f32_to_bf16(acc[nt].f[c] * inv);
  }
}

// ---------------------------------------------------------------------------
// Host-side orchestration
// ---------------------------------------------------------------------------
extern "C" void kernel_launch(void* const* d_in, const int* in_sizes, int n_in,
                              void* d_out, int out_size, void* d_ws, size_t ws_size,
                              hipStream_t stream) {
  const float* query = (const float*)d_in[0];
  const float* key_  = (const float*)d_in[1];
  const float* value = (const float*)d_in[2];
  const float* Wq = (const float*)d_in[3];
  const float* bq = (const float*)d_in[4];
  const float* Wk = (const float*)d_in[5];
  const float* bk = (const float*)d_in[6];
  const float* Wv = (const float*)d_in[7];
  const float* bv = (const float*)d_in[8];
  const float* Wo = (const float*)d_in[9];
  const float* bo = (const float*)d_in[10];

  const int BT = BATCH * SEQ;                // 8192
  const size_t actN = (size_t)BT * DMODEL;   // 8,388,608
  const size_t wN   = (size_t)DMODEL * DMODEL;

  char* ws = (char*)d_ws;
  size_t off = 0;
  auto carve = [&](size_t bytes) {
    void* p = ws + off;
    off += (bytes + 255) & ~(size_t)255;
    return p;
  };
  unsigned short* qb  = (unsigned short*)carve(actN * 2);
  unsigned short* kb  = (unsigned short*)carve(actN * 2);
  unsigned short* vb  = (unsigned short*)carve(actN * 2);
  unsigned short* wqb = (unsigned short*)carve(wN * 2);
  unsigned short* wkb = (unsigned short*)carve(wN * 2);
  unsigned short* wvb = (unsigned short*)carve(wN * 2);
  unsigned short* wob = (unsigned short*)carve(wN * 2);
  unsigned short* Qp  = (unsigned short*)carve(actN * 2);
  unsigned short* Kp  = (unsigned short*)carve(actN * 2);
  unsigned short* Vp  = (unsigned short*)carve(actN * 2);
  unsigned short* ctx = (unsigned short*)carve(actN * 2);

  // 1) fp32 -> bf16
  {
    int blkA = (int)((actN + 1023) / 1024);
    int blkW = (int)((wN + 1023) / 1024);
    cvt_f32_bf16<<<blkA, 256, 0, stream>>>(query, qb, (int)actN);
    cvt_f32_bf16<<<blkA, 256, 0, stream>>>(key_,  kb, (int)actN);
    cvt_f32_bf16<<<blkA, 256, 0, stream>>>(value, vb, (int)actN);
    cvt_f32_bf16<<<blkW, 256, 0, stream>>>(Wq, wqb, (int)wN);
    cvt_f32_bf16<<<blkW, 256, 0, stream>>>(Wk, wkb, (int)wN);
    cvt_f32_bf16<<<blkW, 256, 0, stream>>>(Wv, wvb, (int)wN);
    cvt_f32_bf16<<<blkW, 256, 0, stream>>>(Wo, wob, (int)wN);
  }

  // 2) Q/K/V projections (bf16 out); Q pre-scaled by 1/sqrt(hd) = 0.125
  dim3 gg(DMODEL / 64, BT / 256, 1);
  gemm_bf16<0><<<gg, 256, 0, stream>>>(qb, wqb, bq, Qp, nullptr, 0.125f);
  gemm_bf16<0><<<gg, 256, 0, stream>>>(kb, wkb, bk, Kp, nullptr, 1.0f);
  gemm_bf16<0><<<gg, 256, 0, stream>>>(vb, wvb, bv, Vp, nullptr, 1.0f);

  // 3) flash attention
  dim3 ga(SEQ / 128, NHEAD, BATCH);
  flash_attn<<<ga, 256, 0, stream>>>(Qp, Kp, Vp, ctx);

  // 4) output projection (f32 out)
  gemm_bf16<1><<<gg, 256, 0, stream>>>(ctx, wob, bo, nullptr, (float*)d_out, 1.0f);
}
